// WeatherTrackerSiamTransformer_31817117729541
// MI455X (gfx1250) — matrix operand order repaired and checked by reading, not compiled, not run on hardware
//
#include <hip/hip_runtime.h>

typedef __attribute__((ext_vector_type(16))) _Float16 v16h;
typedef __attribute__((ext_vector_type(8)))  float    v8f;

#define B_     64
#define C_     256
#define TH_    16     // template H
#define TW_    16     // template W
#define SH_    64     // search H
#define SW_    64     // search W
#define OH_    49
#define OW_    49
#define CCHUNK 32
#define IPB    4      // i' rows per block
#define NTHR   512    // 16 waves

// ---------------------------------------------------------------------------
// CDNA5 async global->LDS copy (16B per lane), tracked by ASYNCcnt.
// VDST lane value = LDS byte address; flat shared pointers carry the LDS
// offset in their low 32 bits.
// ---------------------------------------------------------------------------
__device__ __forceinline__ void async_copy_b128(const void* gptr, void* lptr) {
  unsigned l = (unsigned)(uintptr_t)lptr;
  asm volatile("global_load_async_to_lds_b128 %0, %1, off"
               :: "v"(l), "v"(gptr) : "memory");
}

// ---------------------------------------------------------------------------
// 1a) Normalize template over channel dim, convert to f16.
// ---------------------------------------------------------------------------
__global__ __launch_bounds__(256) void norm_template_kernel(
    const float* __restrict__ t, _Float16* __restrict__ t16) {
  int tid = blockIdx.x * blockDim.x + threadIdx.x;      // 0 .. B*256-1
  int b   = tid >> 8;
  int pos = tid & 255;                                  // h*16 + w
  const float* base = t + (size_t)b * C_ * 256 + pos;
  float ss = 0.f;
  for (int c = 0; c < C_; ++c) {
    float v = base[c * 256];
    ss += v * v;
  }
  float inv = 1.f / (sqrtf(ss) + 1e-8f);
  _Float16* obase = t16 + (size_t)b * C_ * 256 + pos;
  for (int c = 0; c < C_; ++c)
    obase[c * 256] = (_Float16)(base[c * 256] * inv);
}

// ---------------------------------------------------------------------------
// 1b) Normalize search, convert to f16.
// ---------------------------------------------------------------------------
__global__ __launch_bounds__(256) void norm_search_kernel(
    const float* __restrict__ s, _Float16* __restrict__ s16) {
  int tid = blockIdx.x * blockDim.x + threadIdx.x;      // 0 .. B*4096-1
  int b   = tid >> 12;
  int pos = tid & 4095;                                 // y*64 + x
  const float* base = s + (size_t)b * C_ * 4096 + pos;
  float ss = 0.f;
  for (int c = 0; c < C_; ++c) {
    float v = base[c * 4096];
    ss += v * v;
  }
  float inv = 1.f / (sqrtf(ss) + 1e-8f);
  _Float16* obase = s16 + (size_t)b * C_ * 4096 + pos;
  for (int c = 0; c < C_; ++c)
    obase[c * 4096] = (_Float16)(base[c * 4096] * inv);
}

// ---------------------------------------------------------------------------
// Async-stage one 32-channel chunk: t (32x256 halves) + s (4 rows x 32c x 64
// halves). Exactly 4 async b128 ops per thread (2 t + 2 s) -> uniform
// ASYNCcnt across the wave, so waits use a fixed immediate.
// ---------------------------------------------------------------------------
__device__ __forceinline__ void stage_chunk_async(
    const _Float16* __restrict__ t16, const _Float16* __restrict__ s16,
    int b, int c0, int ip0, _Float16* t_dst, _Float16* s_dst, int tid) {
  const _Float16* gt = t16 + (size_t)(b * C_ + c0) * 256;
  #pragma unroll
  for (int q = tid; q < 1024; q += NTHR)                 // 2 iterations
    async_copy_b128(gt + q * 8, t_dst + q * 8);
  #pragma unroll
  for (int q = tid; q < 1024; q += NTHR) {               // 2 iterations
    int ipl = q >> 8;                                    // 0..3
    int rem = q & 255;
    int r   = rem >> 3;                                  // channel 0..31
    int ch  = rem & 7;                                   // 16B chunk in row
    const _Float16* gs =
        s16 + ((size_t)(b * C_ + c0 + r) * SH_ + (ip0 + ipl)) * SW_ + ch * 8;
    async_copy_b128(gs, s_dst + ((ipl * CCHUNK) + r) * 80 + ch * 8);
  }
}

// ---------------------------------------------------------------------------
// 2) WMMA correlation with double-buffered async LDS staging.
//    Block = 512 threads = 16 waves: wave -> (i'_local = w>>2, j-tile = w&3).
//    P[b, i', h, j] = sum_{c,w} s16[b,c,i',j+w] * t16[b,c,h,w]
// ---------------------------------------------------------------------------
__global__ __launch_bounds__(NTHR) void corr_wmma_kernel(
    const _Float16* __restrict__ t16, const _Float16* __restrict__ s16,
    float* __restrict__ P) {
  const int ip0 = blockIdx.x * IPB;   // first i' of this block
  const int b   = blockIdx.y;

  __shared__ _Float16 t_sh[2][CCHUNK * TH_ * TW_];     // [buf][cc][h][w] 16KB ea
  __shared__ _Float16 s_sh[2][IPB * CCHUNK * 80];      // [buf][ipl][cc][80] 20KB ea

  const int tid  = threadIdx.x;
  const int lane = tid & 31;
  const int wave = tid >> 5;
  const int ipl  = wave >> 2;                          // 0..3
  const int j0   = (wave & 3) * 16;                    // j-tile origin
  const int m    = lane & 15;
  const int hi   = lane >> 4;
  const int woff = hi * 8;
  const int j    = j0 + m;

  // Zero the pad columns (64..79) of every s row once; async loads never
  // write them, so padded j-tiles read zeros forever after.
  if (tid < 2 * IPB * CCHUNK) {                        // 256 rows
    int buf = tid >> 7, row = tid & 127;
    uint4 z = {0u, 0u, 0u, 0u};
    uint4* zp = reinterpret_cast<uint4*>(&s_sh[buf][row * 80 + 64]);
    zp[0] = z; zp[1] = z;
  }

  // Prologue: kick off chunk 0.
  stage_chunk_async(t16, s16, b, 0, ip0, t_sh[0], s_sh[0], tid);

  v8f acc = {};

  for (int k = 0; k < C_ / CCHUNK; ++k) {              // 8 chunks
    const int buf = k & 1;
    if (k + 1 < C_ / CCHUNK) {
      // Issue next chunk into the other buffer, then wait for the current
      // chunk's 4 ops (ours are in-order; 4 newer ones may stay in flight).
      stage_chunk_async(t16, s16, b, (k + 1) * CCHUNK, ip0,
                        t_sh[buf ^ 1], s_sh[buf ^ 1], tid);
      asm volatile("s_wait_asynccnt 0x4" ::: "memory");
    } else {
      asm volatile("s_wait_asynccnt 0x0" ::: "memory");
    }
    __syncthreads();                                   // publish LDS writes

    const _Float16* sbase = &s_sh[buf][ipl * CCHUNK * 80];
    #pragma unroll
    for (int kk = 0; kk < 16; ++kk) {
      const int cc0 = kk * 2;
      // A gather (16-bit A layout): idx t<8 -> (cc0, w=woff+t),
      //                             idx t>=8 -> (cc0+1, w=woff+t-8)
      const _Float16* srow = sbase + cc0 * 80 + j + woff;
      v16h a;
      #pragma unroll
      for (int t = 0; t < 8; ++t) {
        a[t]     = srow[t];
        a[t + 8] = srow[80 + t];
      }
      // B gather (16-bit B layout): lane col n=m, idx t<8 -> (cc0, w=woff+t),
      // idx t>=8 -> (cc0+1, w=woff+t-8), 16B aligned.
      union { float4 f4[2]; v16h v; } bu;
      const float4* brow = reinterpret_cast<const float4*>(
          &t_sh[buf][(cc0 * TH_ + m) * TW_ + woff]);
      bu.f4[0] = brow[0];
      bu.f4[1] = brow[(TH_ * TW_) / 8];

      acc = __builtin_amdgcn_wmma_f32_16x16x32_f16(
          false, a, false, bu.v, (short)0, acc, false, false);
    }
    __syncthreads();                                   // buffer reuse fence
  }

  // Store P: D layout => lane col n=m is h; VGPR r is M-row j0+r(+8).
  float* prow =
      P + (((size_t)(b * SH_ + ip0 + ipl) * 16) + m) * 64 + j0 + woff;
  reinterpret_cast<float4*>(prow)[0] = make_float4(acc[0], acc[1], acc[2], acc[3]);
  reinterpret_cast<float4*>(prow)[1] = make_float4(acc[4], acc[5], acc[6], acc[7]);
}

// ---------------------------------------------------------------------------
// 3) Combine: sim[b,i,j] = sum_h P[b, i+h, h, j]
// ---------------------------------------------------------------------------
__global__ __launch_bounds__(256) void combine_kernel(
    const float* __restrict__ P, float* __restrict__ out, int n) {
  int tid = blockIdx.x * blockDim.x + threadIdx.x;
  if (tid >= n) return;
  int b   = tid / (OH_ * OW_);
  int rem = tid % (OH_ * OW_);
  int i   = rem / OW_;
  int jj  = rem % OW_;
  float s = 0.f;
  #pragma unroll
  for (int h = 0; h < TH_; ++h)
    s += P[(((size_t)(b * SH_ + (i + h)) * 16) + h) * 64 + jj];
  out[tid] = s;
}

// ---------------------------------------------------------------------------
// launch
// ---------------------------------------------------------------------------
extern "C" void kernel_launch(void* const* d_in, const int* in_sizes, int n_in,
                              void* d_out, int out_size, void* d_ws, size_t ws_size,
                              hipStream_t stream) {
  const float* tmpl = (const float*)d_in[0];   // [64,256,16,16] f32
  const float* srch = (const float*)d_in[1];   // [64,256,64,64] f32
  float* out = (float*)d_out;                  // [64,1,49,49]   f32

  char* ws = (char*)d_ws;
  const size_t t16_bytes = (size_t)B_ * C_ * TH_ * TW_ * 2;   //   8 MB
  const size_t s16_bytes = (size_t)B_ * C_ * SH_ * SW_ * 2;   // 134 MB
  _Float16* t16 = (_Float16*)ws;
  _Float16* s16 = (_Float16*)(ws + t16_bytes);
  float*    P   = (float*)(ws + t16_bytes + s16_bytes);       // 16.8 MB

  norm_template_kernel<<<(B_ * 256) / 256, 256, 0, stream>>>(tmpl, t16);
  norm_search_kernel<<<(B_ * 4096) / 256, 256, 0, stream>>>(srch, s16);

  corr_wmma_kernel<<<dim3(SH_ / IPB, B_), NTHR, 0, stream>>>(t16, s16, P);

  int n = B_ * OH_ * OW_;
  combine_kernel<<<(n + 255) / 256, 256, 0, stream>>>(P, out, n);
}
